// Attention_67808943669336
// MI455X (gfx1250) — matrix operand order repaired and checked by reading, not compiled, not run on hardware
//
#include <hip/hip_runtime.h>
#include <math.h>

// ---------------------------------------------------------------------------
// MI455X (gfx1250) implementation of a llama-style attention block.
// B=1, S=2048, D=4096, H=32, KV=8, HD=128, causal, RoPE, fp32 in/out.
//
// Roofline: ~240 GFLOP vs ~240MB traffic -> compute bound on matrix units.
// Strategy: bf16 V_WMMA_F32_16X16X32_BF16 everywhere, f32 accumulate,
// flash-attention for the S x S score matrix (never materialized).
// GEMM staging uses GLOBAL_LOAD_ASYNC_TO_LDS_B128 (ASYNCcnt-tracked DMA into
// LDS, no VGPR round-trip).
// ---------------------------------------------------------------------------

typedef __bf16 bf16_t;
typedef bf16_t v2bf  __attribute__((ext_vector_type(2)));
typedef bf16_t v4bf  __attribute__((ext_vector_type(4)));
typedef bf16_t v8bf  __attribute__((ext_vector_type(8)));
typedef bf16_t v16bf __attribute__((ext_vector_type(16)));
typedef float  v8f   __attribute__((ext_vector_type(8)));

static constexpr int S  = 2048;
static constexpr int D  = 4096;
static constexpr int H  = 32;
static constexpr int KV = 8;
static constexpr int HD = 128;
static constexpr int F  = H * HD;    // 4096
static constexpr int FK = KV * HD;   // 1024

__device__ __forceinline__ v16bf combine16(v8bf lo, v8bf hi) {
  return __builtin_shufflevector(lo, hi, 0, 1, 2, 3, 4, 5, 6, 7,
                                          8, 9, 10, 11, 12, 13, 14, 15);
}

__device__ __forceinline__ v8f wmma_bf16f32(v16bf a, v16bf b, v8f c) {
  // D = A(16x32 bf16) x B(32x16 bf16) + C(16x16 f32)
  return __builtin_amdgcn_wmma_f32_16x16x32_bf16(
      false, a, false, b, (short)0, c, false, false);
}

// Async DMA of 16 bytes per lane from global memory straight into LDS.
// LDS destination is byte offset lds_off; tracked by ASYNCcnt (not LOADcnt).
__device__ __forceinline__ void async_ld_lds_b128(unsigned lds_off,
                                                  const void* gaddr) {
  asm volatile("global_load_async_to_lds_b128 %0, %1, off"
               :
               : "v"(lds_off), "v"(gaddr)
               : "memory");
}
__device__ __forceinline__ void wait_asynccnt0() {
  asm volatile("s_wait_asynccnt 0" ::: "memory");
}
// Low 32 bits of a generic shared pointer == LDS byte offset (LDS aperture).
__device__ __forceinline__ unsigned lds_offset(const void* p) {
  return (unsigned)(unsigned long long)p;
}

// ---------------------------------------------------------------------------
// f32 -> bf16 elementwise convert (for x)
// ---------------------------------------------------------------------------
__global__ __launch_bounds__(256) void cvt_f32_bf16(
    const float4* __restrict__ in, v4bf* __restrict__ out, int n4) {
  int i = blockIdx.x * blockDim.x + threadIdx.x;
  if (i < n4) {
    float4 v = in[i];
    v4bf o;
    o.x = (bf16_t)v.x; o.y = (bf16_t)v.y; o.z = (bf16_t)v.z; o.w = (bf16_t)v.w;
    out[i] = o;
  }
}

// ---------------------------------------------------------------------------
// f32 [R][C] row-major  ->  bf16 [C][R]   (weights become K-major for WMMA B)
// ---------------------------------------------------------------------------
__global__ __launch_bounds__(256) void transpose_cvt(
    const float* __restrict__ in, bf16_t* __restrict__ out, int R, int C) {
  __shared__ float t[32][33];
  const int tx = threadIdx.x, ty = threadIdx.y;
  const int bx = blockIdx.x, by = blockIdx.y;
#pragma unroll
  for (int i = 0; i < 4; i++)
    t[ty + i * 8][tx] = in[(size_t)(by * 32 + ty + i * 8) * C + bx * 32 + tx];
  __syncthreads();
#pragma unroll
  for (int i = 0; i < 4; i++)
    out[(size_t)(bx * 32 + ty + i * 8) * R + by * 32 + tx] =
        (bf16_t)t[tx][ty + i * 8];
}

// ---------------------------------------------------------------------------
// GEMM: C(MxN) = A(MxK bf16, row-major) * B, with B supplied K-major as
// Bt[N][K].  TRANS_OUT=false -> f32 row-major C.  TRANS_OUT=true -> bf16
// transposed output CbT[N][M] (used for V so attention reads V^T contiguously).
// Block tile 128x128, BK=64, 8 waves of 32x64, WMMA bf16.
// Tiles staged via async global->LDS DMA (ASYNCcnt).
// ---------------------------------------------------------------------------
template <bool TRANS_OUT>
__global__ __launch_bounds__(256) void gemm_bf16(
    const bf16_t* __restrict__ A, const bf16_t* __restrict__ Bt,
    float* __restrict__ Cf, bf16_t* __restrict__ CbT, int M, int N, int K) {
  constexpr int BM = 128, BN = 128, BK = 64, LS = 72;  // padded LDS stride
  __shared__ bf16_t As[BM * LS];
  __shared__ bf16_t Bs[BN * LS];

  const int tid = threadIdx.x;
  const int wv = tid >> 5;
  const int lane = tid & 31;
  const int lane16 = lane & 15;
  const int half = lane >> 4;
  const int wm = wv >> 1;  // 0..3 -> 32-row slab
  const int wn = wv & 1;   // 0..1 -> 64-col slab
  const int bn0 = blockIdx.x * BN;
  const int bm0 = blockIdx.y * BM;

  v8f acc[2][4];
  v8f zero = {};
#pragma unroll
  for (int a = 0; a < 2; a++)
#pragma unroll
    for (int b = 0; b < 4; b++) acc[a][b] = zero;

  // per-thread staging chunk geometry (fixed across K loop)
  for (int kt = 0; kt < K; kt += BK) {
    // ---- stage 128x64 A and B tiles into LDS: 16B async DMA per lane x4 ----
#pragma unroll
    for (int i = 0; i < 4; i++) {
      int c = tid + i * 256;       // 0..1023 chunk id
      int row = c >> 3;            // 0..127
      int off = (c & 7) << 3;      // element offset 0,8,..,56
      async_ld_lds_b128(lds_offset(&As[row * LS + off]),
                        &A[(size_t)(bm0 + row) * K + kt + off]);
      async_ld_lds_b128(lds_offset(&Bs[row * LS + off]),
                        &Bt[(size_t)(bn0 + row) * K + kt + off]);
    }
    if (kt + BK < K) {  // global_prefetch_b8 of next K tile
      __builtin_prefetch(&A[(size_t)(bm0 + (tid >> 1)) * K + kt + BK], 0, 1);
      __builtin_prefetch(&Bt[(size_t)(bn0 + (tid >> 1)) * K + kt + BK], 0, 1);
    }
    wait_asynccnt0();
    __syncthreads();

#pragma unroll
    for (int kc = 0; kc < 2; kc++) {  // two 32-deep WMMA K-chunks
      v16bf af[2];
#pragma unroll
      for (int tm = 0; tm < 2; tm++) {
        // A-fragment: lane = row, two contiguous 8-elem K runs (ISA layout)
        const bf16_t* p =
            &As[(wm * 32 + tm * 16 + lane16) * LS + kc * 32 + half * 8];
        af[tm] = combine16(*(const v8bf*)p, *(const v8bf*)(p + 16));
      }
#pragma unroll
      for (int tn = 0; tn < 4; tn++) {
        // B-fragment: lane = col, two contiguous 8-elem K runs (ISA layout)
        const bf16_t* p =
            &Bs[(wn * 64 + tn * 16 + lane16) * LS + kc * 32 + half * 8];
        v16bf bfr = combine16(*(const v8bf*)p, *(const v8bf*)(p + 16));
#pragma unroll
        for (int tm = 0; tm < 2; tm++)
          acc[tm][tn] = wmma_bf16f32(af[tm], bfr, acc[tm][tn]);
      }
    }
    __syncthreads();
  }

  // ---- epilogue ----
#pragma unroll
  for (int tm = 0; tm < 2; tm++) {
#pragma unroll
    for (int tn = 0; tn < 4; tn++) {
      const int col = bn0 + wn * 64 + tn * 16 + lane16;
      const int rbase = bm0 + wm * 32 + tm * 16 + half * 8;
      if (TRANS_OUT) {
        // lane holds 8 consecutive rows of one column -> one 16B bf16 store
        v8bf o;
#pragma unroll
        for (int v = 0; v < 8; v++) o[v] = (bf16_t)acc[tm][tn][v];
        *(v8bf*)&CbT[(size_t)col * M + rbase] = o;
      } else {
#pragma unroll
        for (int v = 0; v < 8; v++)
          Cf[(size_t)(rbase + v) * N + col] = acc[tm][tn][v];
      }
    }
  }
}

// ---------------------------------------------------------------------------
// RoPE on f32 GEMM output -> bf16 row-major (pairs of consecutive elements)
// ---------------------------------------------------------------------------
__global__ __launch_bounds__(256) void rope_cvt(
    const float* __restrict__ qf, const float* __restrict__ cosb,
    const float* __restrict__ sinb, bf16_t* __restrict__ qb, int nh) {
  const int idx = blockIdx.x * blockDim.x + threadIdx.x;
  const int total = S * nh * (HD / 2);
  if (idx >= total) return;
  const int d2 = idx & 63;
  const int h = (idx >> 6) % nh;
  const int s = idx / (nh * 64);
  const size_t base = ((size_t)s * nh + h) * HD + 2 * d2;
  const float a = qf[base], b = qf[base + 1];
  const float c = cosb[s * 64 + d2], sn = sinb[s * 64 + d2];
  v2bf o;
  o.x = (bf16_t)(a * c - b * sn);
  o.y = (bf16_t)(a * sn + b * c);
  *(v2bf*)&qb[base] = o;
}

// ---------------------------------------------------------------------------
// Flash attention.  One wave per (head, 16-query tile).  Keys in blocks of
// 32 (two 16x16 score tiles -> one 16x32 P A-matrix).  Online softmax in
// exp2 domain; cross-lane reductions via shfl_xor (stay inside 16-lane
// halves, matching the WMMA C layout).  V is read from V^T (K-major).
// ---------------------------------------------------------------------------
__global__ __launch_bounds__(128) void attn_kernel(
    const bf16_t* __restrict__ Qb, const bf16_t* __restrict__ Kb,
    const bf16_t* __restrict__ VbT, bf16_t* __restrict__ attnb) {
  constexpr int PST = 48;  // P staging LDS stride (bf16 elems), 16B aligned
  __shared__ bf16_t Pl[4][16 * PST];

  const int wave = threadIdx.x >> 5;
  const int lane = threadIdx.x & 31;
  const int lane16 = lane & 15;
  const int half = lane >> 4;
  const int w = blockIdx.x * 4 + wave;
  const int head = w >> 7;  // 128 q-tiles per head
  const int qt = w & 127;
  const int kh = head >> 2;  // GQA: 4 q-heads per kv head
  bf16_t* Pw = &Pl[wave][0];

  // ---- Q fragments: 16 rows x 128 (4 A-fragments of 16x32) ----
  v16bf qf[4];
  {
    const int qrow = qt * 16 + lane16;
    const bf16_t* qp0 = Qb + (size_t)qrow * F + head * HD;
#pragma unroll
    for (int kc = 0; kc < 4; kc++) {
      const bf16_t* p = qp0 + kc * 32 + half * 8;
      qf[kc] = combine16(*(const v8bf*)p, *(const v8bf*)(p + 16));
    }
  }

  v8f O[8];
  v8f zero = {};
#pragma unroll
  for (int nc = 0; nc < 8; nc++) O[nc] = zero;
  float m[8], l[8];
#pragma unroll
  for (int v = 0; v < 8; v++) { m[v] = -1e30f; l[v] = 0.0f; }

  const float sc = 0.08838834764831845f * 1.4426950408889634f;  // rsqrt(128)*log2e
  const int nkb = (qt * 16 + 16 + 31) >> 5;  // causal: key blocks of 32

  for (int kb32 = 0; kb32 < nkb; kb32++) {
    const int kb = kb32 * 32;

    // ---- scores: two 16x16 tiles over 128-deep K ----
    v8f s0 = zero, s1 = zero;
#pragma unroll
    for (int kc = 0; kc < 4; kc++) {
      const bf16_t* kp0 =
          Kb + (size_t)(kb + lane16) * FK + kh * HD + kc * 32 + half * 8;
      const bf16_t* kp1 = kp0 + (size_t)16 * FK;
      v16bf b0 = combine16(*(const v8bf*)kp0, *(const v8bf*)(kp0 + 16));
      v16bf b1 = combine16(*(const v8bf*)kp1, *(const v8bf*)(kp1 + 16));
      s0 = wmma_bf16f32(qf[kc], b0, s0);
      s1 = wmma_bf16f32(qf[kc], b1, s1);
    }

    // ---- online softmax (per-row stats; rows = vgpr + 8*half) ----
    const int k0 = kb + lane16;
    const int k1 = kb + 16 + lane16;
    float p0[8], p1[8], alph[8];
#pragma unroll
    for (int v = 0; v < 8; v++) {
      const int qpos = qt * 16 + v + half * 8;
      float a0 = (k0 <= qpos) ? s0[v] * sc : -1e30f;
      float a1 = (k1 <= qpos) ? s1[v] * sc : -1e30f;
      float rm = fmaxf(a0, a1);
      rm = fmaxf(rm, __shfl_xor(rm, 1, 32));
      rm = fmaxf(rm, __shfl_xor(rm, 2, 32));
      rm = fmaxf(rm, __shfl_xor(rm, 4, 32));
      rm = fmaxf(rm, __shfl_xor(rm, 8, 32));
      const float mn = fmaxf(m[v], rm);
      alph[v] = exp2f(m[v] - mn);
      m[v] = mn;
      const float e0 = exp2f(a0 - mn);
      const float e1 = exp2f(a1 - mn);
      p0[v] = e0; p1[v] = e1;
      float rs = e0 + e1;
      rs += __shfl_xor(rs, 1, 32);
      rs += __shfl_xor(rs, 2, 32);
      rs += __shfl_xor(rs, 4, 32);
      rs += __shfl_xor(rs, 8, 32);
      l[v] = l[v] * alph[v] + rs;
    }
#pragma unroll
    for (int nc = 0; nc < 8; nc++)
#pragma unroll
      for (int v = 0; v < 8; v++) O[nc][v] *= alph[v];

    // ---- transpose P (C layout) -> A layout via per-wave LDS ----
#pragma unroll
    for (int v = 0; v < 8; v++) {
      const int r = v + half * 8;
      Pw[r * PST + lane16] = (bf16_t)p0[v];
      Pw[r * PST + 16 + lane16] = (bf16_t)p1[v];
    }
    asm volatile("s_wait_dscnt 0" ::: "memory");  // same-wave LDS RAW
    const bf16_t* pp = Pw + lane16 * PST + half * 8;
    const v16bf pf = combine16(*(const v8bf*)pp, *(const v8bf*)(pp + 16));

    // ---- O += P(16x32) * V(32x128), V read K-major from V^T ----
#pragma unroll
    for (int nc = 0; nc < 8; nc++) {
      const bf16_t* vp =
          VbT + (size_t)(kh * HD + nc * 16 + lane16) * S + kb + half * 8;
      v16bf vf = combine16(*(const v8bf*)vp, *(const v8bf*)(vp + 16));
      O[nc] = wmma_bf16f32(pf, vf, O[nc]);
    }
  }

  // ---- normalize and store bf16 row-major for the output projection ----
#pragma unroll
  for (int v = 0; v < 8; v++) l[v] = 1.0f / l[v];
#pragma unroll
  for (int nc = 0; nc < 8; nc++) {
    const int col = head * HD + nc * 16 + lane16;
#pragma unroll
    for (int v = 0; v < 8; v++) {
      const int row = qt * 16 + v + half * 8;
      attnb[(size_t)row * F + col] = (bf16_t)(O[nc][v] * l[v]);
    }
  }
}

// ---------------------------------------------------------------------------
// launch
// ---------------------------------------------------------------------------
extern "C" void kernel_launch(void* const* d_in, const int* in_sizes, int n_in,
                              void* d_out, int out_size, void* d_ws,
                              size_t ws_size, hipStream_t stream) {
  const float* x = (const float*)d_in[0];
  const float* fcos = (const float*)d_in[1];
  const float* fsin = (const float*)d_in[2];
  // d_in[3] = mask (causal computed inline), d_in[8] = start_pos (0)
  const float* wq = (const float*)d_in[4];
  const float* wk = (const float*)d_in[5];
  const float* wv = (const float*)d_in[6];
  const float* wo = (const float*)d_in[7];
  float* out = (float*)d_out;

  char* ws = (char*)d_ws;
  size_t off = 0;
  auto take = [&](size_t bytes) {
    void* p = ws + off;
    off += (bytes + 255) & ~(size_t)255;
    return p;
  };
  bf16_t* xb   = (bf16_t*)take((size_t)S * D * 2);    // x bf16
  bf16_t* wqT  = (bf16_t*)take((size_t)F * D * 2);    // [N][K]
  bf16_t* wkT  = (bf16_t*)take((size_t)FK * D * 2);
  bf16_t* wvT  = (bf16_t*)take((size_t)FK * D * 2);
  bf16_t* woT  = (bf16_t*)take((size_t)D * F * 2);
  float*  Qf   = (float*)take((size_t)S * F * 4);     // pre-RoPE Q
  float*  Kf   = (float*)take((size_t)S * FK * 4);    // pre-RoPE K
  bf16_t* Qb   = (bf16_t*)take((size_t)S * F * 2);
  bf16_t* Kb   = (bf16_t*)take((size_t)S * FK * 2);
  bf16_t* VbT  = (bf16_t*)take((size_t)FK * S * 2);   // V transposed [HD*KV][S]
  bf16_t* attnb = (bf16_t*)take((size_t)S * F * 2);
  (void)ws_size;

  // 1) convert x to bf16
  {
    int n4 = S * D / 4;
    cvt_f32_bf16<<<n4 / 256, 256, 0, stream>>>((const float4*)x, (v4bf*)xb, n4);
  }
  // 2) transpose-convert weights to K-major bf16
  {
    dim3 blk(32, 8);
    transpose_cvt<<<dim3(F / 32, D / 32), blk, 0, stream>>>(wq, wqT, D, F);
    transpose_cvt<<<dim3(FK / 32, D / 32), blk, 0, stream>>>(wk, wkT, D, FK);
    transpose_cvt<<<dim3(FK / 32, D / 32), blk, 0, stream>>>(wv, wvT, D, FK);
    transpose_cvt<<<dim3(D / 32, F / 32), blk, 0, stream>>>(wo, woT, F, D);
  }
  // 3) projections
  gemm_bf16<false><<<dim3(F / 128, S / 128), 256, 0, stream>>>(
      xb, wqT, Qf, nullptr, S, F, D);
  gemm_bf16<false><<<dim3(FK / 128, S / 128), 256, 0, stream>>>(
      xb, wkT, Kf, nullptr, S, FK, D);
  gemm_bf16<true><<<dim3(FK / 128, S / 128), 256, 0, stream>>>(
      xb, wvT, nullptr, VbT, S, FK, D);
  // 4) RoPE + bf16
  rope_cvt<<<(S * H * 64) / 256, 256, 0, stream>>>(Qf, fcos, fsin, Qb, H);
  rope_cvt<<<(S * KV * 64) / 256, 256, 0, stream>>>(Kf, fcos, fsin, Kb, KV);
  // 5) flash attention: 32 heads x 128 q-tiles, 4 waves/block
  attn_kernel<<<(H * (S / 16)) / 4, 128, 0, stream>>>(Qb, Kb, VbT, attnb);
  // 6) output projection -> f32
  gemm_bf16<false><<<dim3(D / 128, S / 128), 256, 0, stream>>>(
      attnb, woT, out, nullptr, S, D, F);
}
